// MultiHeadAttention_25512105738403
// MI455X (gfx1250) — matrix operand order repaired and checked by reading, not compiled, not run on hardware
//
#include <hip/hip_runtime.h>
#include <math.h>

// ---------------- problem constants (fixed by the reference) ----------------
#define BB    2
#define SS    2048
#define MMDIM 1024
#define HH    16
#define HKV   4
#define DD    64
#define BSROWS (BB * SS)   // 4096

typedef __attribute__((ext_vector_type(16))) __bf16 bf16x16;
typedef __attribute__((ext_vector_type(8)))  __bf16 bf16x8;
typedef __attribute__((ext_vector_type(8)))  float  f32x8;

static __device__ __forceinline__ bf16x16 cat8(bf16x8 lo, bf16x8 hi) {
  return __builtin_shufflevector(lo, hi, 0,1,2,3,4,5,6,7,8,9,10,11,12,13,14,15);
}
static __device__ __forceinline__ f32x8 wmma_bf16(bf16x16 a, bf16x16 b, f32x8 c) {
  // v_wmma_f32_16x16x32_bf16  D = A*B + C  (fp32 accumulate)
  return __builtin_amdgcn_wmma_f32_16x16x32_bf16(false, a, false, b, (short)0, c, false, false);
}
// 32-bit LDS offset of a generic pointer to __shared__ memory (for async VDST)
static __device__ __forceinline__ unsigned lds_addr(const void* p) {
  return (unsigned)(unsigned long long)(__attribute__((address_space(3))) const char*)p;
}
// global -> LDS async copy, 16 bytes/lane; INST_OFFSET applies to both sides
#define ASYNC_B128(ldsoff, gaddr)                                              \
  asm volatile("global_load_async_to_lds_b128 %0, %1, off"                     \
               :: "v"(ldsoff), "v"(gaddr) : "memory")
#define ASYNC_B128_OFF(ldsoff, gaddr, imm)                                     \
  asm volatile("global_load_async_to_lds_b128 %0, %1, off offset:" #imm        \
               :: "v"(ldsoff), "v"(gaddr) : "memory")

// pack two fp32 into a dword of two bf16
static __device__ __forceinline__ unsigned pkbf(float a, float b) {
  union { __bf16 h[2]; unsigned u; } u;
  u.h[0] = (__bf16)a; u.h[1] = (__bf16)b;
  return u.u;
}

// ============================================================================
// Prep: fp32 -> bf16 row cast
// ============================================================================
__global__ __launch_bounds__(256) void cast_f32_bf16(
    const float* __restrict__ in, __bf16* __restrict__ outp) {
  const int base = (blockIdx.x * 256 + threadIdx.x) * 8;
  bf16x8 v;
#pragma unroll
  for (int j = 0; j < 8; ++j) v[j] = (__bf16)in[base + j];
  *(bf16x8*)(outp + base) = v;
}

// ============================================================================
// Prep: W[K][NC] fp32 -> WT[NC][K] bf16 (one-time, weights are tiny)
// ============================================================================
__global__ __launch_bounds__(256) void transpose_cast(
    const float* __restrict__ W, __bf16* __restrict__ WT, int K, int NC) {
  const int idx = blockIdx.x * 256 + threadIdx.x;   // over NC*K outputs
  const int n = idx / K, k = idx - n * K;
  WT[idx] = (__bf16)W[(size_t)k * NC + n];
}

// ============================================================================
// GEMM: C[MR x NC](fp32) = A[MR x K](bf16) * BT[NC x K](bf16, pre-transposed)
// Workgroup tile 64x128, 4 waves, wave tile 32x64. K-step 32.
// LDS tiles double-buffered, filled by GLOBAL_LOAD_ASYNC_TO_LDS_B128.
// ============================================================================
template<int NC, int K>
__global__ __launch_bounds__(128) void gemm_bf16wmma(
    const __bf16* __restrict__ A, const __bf16* __restrict__ BT,
    float* __restrict__ C) {
  __shared__ __align__(128) __bf16 As[2][64][32];     // 8 KB
  __shared__ __align__(128) __bf16 BsT[2][128][32];   // 16 KB

  const int tid  = threadIdx.x;
  const int lane = tid & 31;
  const int w    = tid >> 5;
  const int lh   = lane >> 4;
  const int ln   = lane & 15;
  const int rowbase = blockIdx.y * 64;
  const int colbase = blockIdx.x * 128;
  const int wrow = (w >> 1) * 32;
  const int wcol = (w & 1) * 64;

  const int ar = tid >> 1;
  const int ac = (tid & 1) * 16;
  const __bf16* gA = A + (size_t)(rowbase + ar) * K + ac;
  const __bf16* gB = BT + (size_t)(colbase + tid) * K;

  f32x8 acc[2][4] = {};

  auto issue = [&](int buf, int k0) {
    const unsigned la = lds_addr(&As[buf][ar][ac]);
    const unsigned lb = lds_addr(&BsT[buf][tid][0]);
    const unsigned long long a64 = (unsigned long long)(gA + k0);
    const unsigned long long b64 = (unsigned long long)(gB + k0);
    ASYNC_B128(la, a64);
    ASYNC_B128_OFF(la, a64, 16);
    ASYNC_B128(lb, b64);
    ASYNC_B128_OFF(lb, b64, 16);
    ASYNC_B128_OFF(lb, b64, 32);
    ASYNC_B128_OFF(lb, b64, 48);
  };

  issue(0, 0);
  for (int k0 = 0; k0 < K; k0 += 32) {
    const int cur = (k0 >> 5) & 1;
    if (k0 + 32 < K) {
      issue(cur ^ 1, k0 + 32);                          // overlap next tile DMA
      asm volatile("s_wait_asynccnt 0x6" ::: "memory"); // cur tile landed
    } else {
      asm volatile("s_wait_asynccnt 0x0" ::: "memory");
    }
    __syncthreads();

    bf16x16 aop[2], bop[4];
#pragma unroll
    for (int mt = 0; mt < 2; ++mt) {
      const __bf16* p = &As[cur][wrow + mt * 16 + ln][lh * 16];
      aop[mt] = cat8(*(const bf16x8*)p, *(const bf16x8*)(p + 8));
    }
#pragma unroll
    for (int nt = 0; nt < 4; ++nt)
      bop[nt] = *(const bf16x16*)&BsT[cur][wcol + nt * 16 + ln][lh * 16];

#pragma unroll
    for (int mt = 0; mt < 2; ++mt)
#pragma unroll
      for (int nt = 0; nt < 4; ++nt)
        acc[mt][nt] = wmma_bf16(aop[mt], bop[nt], acc[mt][nt]);

    __syncthreads();   // compute done before next iteration's DMA overwrites
  }

#pragma unroll
  for (int mt = 0; mt < 2; ++mt)
#pragma unroll
    for (int nt = 0; nt < 4; ++nt) {
      float* cp = C + (size_t)(rowbase + wrow + mt * 16 + 8 * lh) * NC
                    + colbase + wcol + nt * 16 + ln;
#pragma unroll
      for (int i = 0; i < 8; ++i) cp[(size_t)i * NC] = acc[mt][nt][i];
    }
}

// ============================================================================
// RMS-norm + 2D RoPE (q,k -> bf16) and V -> transposed bf16 [b][hkv][d][s].
// One wave per 64-elem head row; 24 rows per (b,s).
// ============================================================================
__global__ __launch_bounds__(256) void norm_rope_cast(
    const float* __restrict__ qf, const float* __restrict__ kf,
    const float* __restrict__ vf,
    const int* __restrict__ q_pos, const int* __restrict__ k_pos,
    __bf16* __restrict__ qb, __bf16* __restrict__ kb, __bf16* __restrict__ vt) {
  const int row  = blockIdx.x * 8 + (threadIdx.x >> 5);
  const int lane = threadIdx.x & 31;
  const int bs = row / 24;
  const int t  = row - bs * 24;
  const int half = lane >> 4;       // which D/2 half (x/y rope component)
  const int i    = lane & 15;       // rotation pair index within half
  const int e1 = half * 32 + i;
  const int e2 = e1 + 16;

  if (t >= HH + HKV) {              // V: transpose + cast only
    const int hv = t - HH - HKV;
    const float* src = vf + ((size_t)bs * HKV + hv) * DD;
    const int b = bs >> 11, sIdx = bs & (SS - 1);
    __bf16* dst = vt + ((size_t)(b * HKV + hv) * DD) * SS + sIdx;
    dst[(size_t)e1 * SS] = (__bf16)src[e1];
    dst[(size_t)e2 * SS] = (__bf16)src[e2];
    return;
  }

  const float* src; __bf16* dst; const int* pos;
  if (t < HH) {
    src = qf + ((size_t)bs * HH + t) * DD;
    dst = qb + ((size_t)bs * HH + t) * DD;
    pos = q_pos + (size_t)bs * 2;
  } else {
    src = kf + ((size_t)bs * HKV + (t - HH)) * DD;
    dst = kb + ((size_t)bs * HKV + (t - HH)) * DD;
    pos = k_pos + (size_t)bs * 2;
  }

  float x1 = src[e1], x2 = src[e2];
  float ss = x1 * x1 + x2 * x2;
#pragma unroll
  for (int m = 1; m < 32; m <<= 1) ss += __shfl_xor(ss, m, 32);
  const float r = rsqrtf(ss * (1.0f / 64.0f) + 1e-6f);
  x1 *= r; x2 *= r;
  const float p = (float)pos[half];
  const float f = p * __expf(-(float)i * (9.21034037197618f / 16.0f));
  float sn, cs;
  __sincosf(f, &sn, &cs);
  dst[e1] = (__bf16)(x1 * cs - x2 * sn);
  dst[e2] = (__bf16)(x2 * cs + x1 * sn);
}

// ============================================================================
// Flash attention, transposed formulation (no LDS, no barriers):
//   S^T = K * Q^T   (keys = M rows from global, queries = N lanes)
//   O^T = V^T * P^T (V^T rows direct from pre-transposed V)
// One-pass softmax without max subtraction: RMS-normed q,k give |s| <= 8,
// so exp(s) and its 2048-term sum are far from fp32 overflow. Denominator is
// one running scalar per lane (all 8 acc rows belong to ONE query), reduced
// with a single shfl_xor(16) at the end.
// grid (S/64, H, B), 128 threads = 4 independent waves x 16 queries.
// ============================================================================
__global__ __launch_bounds__(128) void flash_attn(
    const __bf16* __restrict__ qb, const __bf16* __restrict__ kbuf,
    const __bf16* __restrict__ vt, __bf16* __restrict__ outp) {
  const int b  = blockIdx.z, h = blockIdx.y, hk = h >> 2;  // GQA: 4 q per kv
  const int w    = threadIdx.x >> 5;
  const int lane = threadIdx.x & 31;
  const int lh = lane >> 4, ln = lane & 15;
  const int q0 = blockIdx.x * 64 + w * 16;   // wave's query base

  // Q as B operand (32x16): lane = query ln; K = d chunk c*32 + lh*16 + {0..15}
  bf16x16 bq[2];
  {
    const __bf16* qp = qb + (((size_t)(b * SS + q0 + ln)) * HH + h) * DD;
#pragma unroll
    for (int c = 0; c < 2; ++c)
      bq[c] = *(const bf16x16*)(qp + c * 32 + lh * 16);
  }

  // V^T row pointers per d-tile: A operand lane = d row (dt*16 + ln)
  const __bf16* vrow[4];
#pragma unroll
  for (int dt = 0; dt < 4; ++dt)
    vrow[dt] = vt + ((size_t)(b * HKV + hk) * DD + dt * 16 + ln) * SS;

  f32x8 ot[4] = {};
  float lsum = 0.0f;

  for (int kb0 = 0; kb0 < SS; kb0 += 32) {
    if (kb0 + 32 < SS) {  // prefetch next K/V blocks -> global_prefetch_b8
      __builtin_prefetch(
          kbuf + (((size_t)(b * SS + kb0 + 32 + lane)) * HKV + hk) * DD, 0, 1);
      __builtin_prefetch(vrow[lane & 3] + kb0 + 32, 0, 1);
    }

    // S^T tiles: st[kt] = K(16x64) * Q^T, kt = key sub-tile
    f32x8 st[2];
#pragma unroll
    for (int kt = 0; kt < 2; ++kt) {
      // A operand: lane = key row (kb0 + kt*16 + ln); K chunks over d
      const __bf16* kp =
          kbuf + (((size_t)(b * SS + kb0 + kt * 16 + ln)) * HKV + hk) * DD;
      f32x8 s = {};
#pragma unroll
      for (int c = 0; c < 2; ++c) {
        const __bf16* p = kp + c * 32 + lh * 16;
        s = wmma_bf16(cat8(*(const bf16x8*)p, *(const bf16x8*)(p + 8)),
                      bq[c], s);
      }
      st[kt] = s;
    }

    // p = exp(s/8); per-lane denominator accumulation (no cross-lane work)
#pragma unroll
    for (int kt = 0; kt < 2; ++kt)
#pragma unroll
      for (int i = 0; i < 8; ++i) {
        const float p = __expf(st[kt][i] * 0.125f);
        st[kt][i] = p;
        lsum += p;
      }

    // Build P^T B-operand (32 keys x 16 queries) from S^T C-layout tiles.
    // Lane half lh needs keys lh*16+{0..15} of query ln: own half supplies 8,
    // the xor-16 partner supplies the other 8 (packed bf16 shuffles).
    unsigned pk0[4], pk1[4], bp[8];
#pragma unroll
    for (int j = 0; j < 4; ++j) {
      pk0[j] = pkbf(st[0][2 * j], st[0][2 * j + 1]);
      pk1[j] = pkbf(st[1][2 * j], st[1][2 * j + 1]);
    }
#pragma unroll
    for (int j = 0; j < 4; ++j) {
      const unsigned sw0 = __shfl_xor(pk0[j], 16, 32);
      const unsigned sw1 = __shfl_xor(pk1[j], 16, 32);
      bp[j]     = lh ? sw1 : pk0[j];   // keys (lh*16) + {0..7}
      bp[4 + j] = lh ? pk1[j] : sw0;   // keys (lh*16) + {8..15}
    }
    union { unsigned u[8]; bf16x16 v; } pb;
#pragma unroll
    for (int j = 0; j < 8; ++j) pb.u[j] = bp[j];

    // O^T += V^T(16d x 32k) * P^T(32k x 16q), per d-tile
#pragma unroll
    for (int dt = 0; dt < 4; ++dt) {
      const __bf16* p = vrow[dt] + kb0 + lh * 16;
      ot[dt] = wmma_bf16(cat8(*(const bf16x8*)p, *(const bf16x8*)(p + 8)),
                         pb.v, ot[dt]);
    }
  }

  // Softmax denominator: keys split across the two lane halves only
  const float inv = 1.0f / (lsum + __shfl_xor(lsum, 16, 32));

  // O^T C-layout: VGPR i -> d = dt*16 + 8*lh + i (contiguous!), lane = query
  __bf16* op = outp + (((size_t)(b * SS + q0 + ln)) * HH + h) * DD;
#pragma unroll
  for (int dt = 0; dt < 4; ++dt) {
    bf16x8 vv;
#pragma unroll
    for (int i = 0; i < 8; ++i) vv[i] = (__bf16)(ot[dt][i] * inv);
    *(bf16x8*)(op + dt * 16 + 8 * lh) = vv;
  }
}

// ============================================================================
extern "C" void kernel_launch(void* const* d_in, const int* in_sizes, int n_in,
                              void* d_out, int out_size, void* d_ws, size_t ws_size,
                              hipStream_t stream) {
  (void)in_sizes; (void)n_in; (void)out_size; (void)ws_size;
  const float* x    = (const float*)d_in[0];   // [B,S,M]
  const int*   qpos = (const int*)d_in[1];     // [B,S,2]
  const int*   kpos = (const int*)d_in[2];     // [B,S,2]
  const float* Wq   = (const float*)d_in[3];   // [M,H*D]   = [1024,1024]
  const float* Wk   = (const float*)d_in[4];   // [M,Hkv*D] = [1024,256]
  const float* Wv   = (const float*)d_in[5];   // [M,Hkv*D] = [1024,256]
  const float* Wo   = (const float*)d_in[6];   // [H*D,M]   = [1024,1024]
  float* out = (float*)d_out;                  // [B,S,M]

  char* ws = (char*)d_ws;
  size_t off = 0;
  auto carve = [&](size_t bytes) -> char* {
    char* p = ws + off;
    off += (bytes + 255) & ~(size_t)255;
    return p;
  };
  float*  qf   = (float*)carve((size_t)BSROWS * HH  * DD * 4);   // 16 MB
  float*  kf   = (float*)carve((size_t)BSROWS * HKV * DD * 4);   //  4 MB
  float*  vf   = (float*)carve((size_t)BSROWS * HKV * DD * 4);   //  4 MB
  __bf16* qbv  = (__bf16*)carve((size_t)BSROWS * HH  * DD * 2);  //  8 MB
  __bf16* kbv  = (__bf16*)carve((size_t)BSROWS * HKV * DD * 2);  //  2 MB
  __bf16* vtb  = (__bf16*)carve((size_t)BSROWS * HKV * DD * 2);  //  2 MB (V^T)
  __bf16* xb   = (__bf16*)carve((size_t)BSROWS * MMDIM * 2);     //  8 MB
  __bf16* WqT  = (__bf16*)carve((size_t)MMDIM * HH  * DD * 2);   //  2 MB
  __bf16* WkT  = (__bf16*)carve((size_t)MMDIM * HKV * DD * 2);   // .5 MB
  __bf16* WvT  = (__bf16*)carve((size_t)MMDIM * HKV * DD * 2);   // .5 MB
  __bf16* WoT  = (__bf16*)carve((size_t)MMDIM * HH  * DD * 2);   //  2 MB
  __bf16* attnb = (__bf16*)qf;  // qf dead after norm_rope_cast; reuse (bf16)

  const dim3 blk128(128), blk256(256);
  // One-time prep: cast x, transpose+cast weights (all bf16)
  cast_f32_bf16<<<dim3(BSROWS * MMDIM / (256 * 8)), blk256, 0, stream>>>(x, xb);
  transpose_cast<<<dim3(MMDIM * 1024 / 256), blk256, 0, stream>>>(Wq, WqT, MMDIM, 1024);
  transpose_cast<<<dim3(MMDIM * 256  / 256), blk256, 0, stream>>>(Wk, WkT, MMDIM, 256);
  transpose_cast<<<dim3(MMDIM * 256  / 256), blk256, 0, stream>>>(Wv, WvT, MMDIM, 256);
  transpose_cast<<<dim3(MMDIM * 1024 / 256), blk256, 0, stream>>>(Wo, WoT, 1024, MMDIM);

  // Q/K/V projections: xb[4096,1024] @ W^T
  gemm_bf16wmma<1024, 1024><<<dim3(1024 / 128, BSROWS / 64), blk128, 0, stream>>>(xb, WqT, qf);
  gemm_bf16wmma< 256, 1024><<<dim3( 256 / 128, BSROWS / 64), blk128, 0, stream>>>(xb, WkT, kf);
  gemm_bf16wmma< 256, 1024><<<dim3( 256 / 128, BSROWS / 64), blk128, 0, stream>>>(xb, WvT, vf);
  // RMS norm + 2D RoPE + bf16 cast (+ V transpose)
  norm_rope_cast<<<dim3(BSROWS * 24 / 8), blk256, 0, stream>>>(
      qf, kf, vf, qpos, kpos, qbv, kbv, vtb);
  // Flash attention (transposed, bf16 out)
  flash_attn<<<dim3(SS / 64, HH, BB), blk128, 0, stream>>>(qbv, kbv, vtb, attnb);
  // Output projection: attn[4096,1024] @ Wo^T -> fp32 out
  gemm_bf16wmma<1024, 1024><<<dim3(MMDIM / 128, BSROWS / 64), blk128, 0, stream>>>(attnb, WoT, out);
}
